// GCNEncoder_63290638073922
// MI455X (gfx1250) — compile-verified, matrix operand order checked by reading, not run on hardware
//
#include <hip/hip_runtime.h>

#define NPTS 8192
#define BATCH 2
#define KNN 10

typedef __attribute__((ext_vector_type(16))) _Float16 v16h;
typedef __attribute__((ext_vector_type(8)))  float    v8f;
typedef __attribute__((__vector_size__(16))) int      i32x4;

__device__ __forceinline__ unsigned encf(float v) {
    unsigned u = __float_as_uint(v);
    return (u & 0x80000000u) ? ~u : (u | 0x80000000u);
}
__device__ __forceinline__ float decf(unsigned u) {
    return (u & 0x80000000u) ? __uint_as_float(u ^ 0x80000000u)
                             : __uint_as_float(~u);
}

// ---------------------------------------------------------------------------
// Fold BN into per-out-channel scale/bias; convert W to f16 padded [Mpad][Kpad]
// y = gamma*(W x + b - mean)*rsqrt(var+eps) + beta  ==  s*(W x) + t
// Padded rows get s=t=0 so padded channels are exactly 0 after the epilogue.
// Column remap (for the c3 concat layout [x1 padded to 48 | x2]):
//   padded col kk maps to source k = kk            (kk <  ksplit)
//                                   = kk - kshift  (kk >= ksplit + kshift)
//                                   = zero         (otherwise / out of range)
// ---------------------------------------------------------------------------
__global__ void prep_layer_kernel(const float* __restrict__ w,
                                  const float* __restrict__ bias,
                                  const float* __restrict__ gamma,
                                  const float* __restrict__ beta,
                                  const float* __restrict__ mean,
                                  const float* __restrict__ var,
                                  int M, int Kin, int Mpad, int Kpad,
                                  int ksplit, int kshift,
                                  _Float16* __restrict__ wout,
                                  float* __restrict__ sout,
                                  float* __restrict__ tout) {
    int stride = gridDim.x * blockDim.x;
    int tid = blockIdx.x * blockDim.x + threadIdx.x;
    int total = Mpad * Kpad;
    for (int i = tid; i < total; i += stride) {
        int m = i / Kpad, kk = i - m * Kpad;
        int k = -1;
        if (kk < ksplit) k = kk;
        else if (kk >= ksplit + kshift) k = kk - kshift;
        float v = (m < M && k >= 0 && k < Kin) ? w[m * Kin + k] : 0.0f;
        wout[i] = (_Float16)v;
    }
    for (int m = tid; m < Mpad; m += stride) {
        float s = 0.0f, t = 0.0f;
        if (m < M) {
            float rs = rsqrtf(var[m] + 1e-5f);
            s = gamma[m] * rs;
            t = s * (bias[m] - mean[m]) + beta[m];
        }
        sout[m] = s;
        tout[m] = t;
    }
}

// ---------------------------------------------------------------------------
// Brute-force KNN (top-K by neg squared distance, self included — matches
// jax.lax.top_k on neg_d whose diagonal is 0 == max). One query per thread,
// neighbors staged through LDS (f32) in 128-wide tiles; inner reads are
// wave-broadcast. Next tile is prefetched (global_prefetch_b8).
//   STRIDE == 0 : input is f32 channel-major  [b][C][N]
//   STRIDE  > 0 : input is f16 point-major    [b][n][STRIDE]
// ---------------------------------------------------------------------------
template <int C, int STRIDE>
__global__ __launch_bounds__(128) void knn_kernel(const void* __restrict__ xv,
                                                  int* __restrict__ idx) {
    constexpr int CP = C + 1;  // pad stride to dodge bank conflicts
    __shared__ float tile[128 * CP];
    const int b = blockIdx.y;
    const int n = blockIdx.x * 128 + threadIdx.x;

    float q[C];
    if (STRIDE == 0) {
        const float* xb = (const float*)xv + (size_t)b * C * NPTS;
#pragma unroll
        for (int c = 0; c < C; ++c) q[c] = xb[c * NPTS + n];
    } else {
        const _Float16* xr =
            (const _Float16*)xv + ((size_t)b * NPTS + n) * (STRIDE ? STRIDE : 1);
#pragma unroll
        for (int c = 0; c < C; ++c) q[c] = (float)xr[c];
    }

    float bd[KNN];
    int bi[KNN];
#pragma unroll
    for (int k = 0; k < KNN; ++k) { bd[k] = -3.4e38f; bi[k] = 0; }

    for (int j0 = 0; j0 < NPTS; j0 += 128) {
        __syncthreads();
        if (STRIDE == 0) {
            const float* xb = (const float*)xv + (size_t)b * C * NPTS;
#pragma unroll
            for (int c = 0; c < C; ++c)
                tile[threadIdx.x * CP + c] = xb[c * NPTS + j0 + threadIdx.x];
            if (j0 + 128 < NPTS)
                __builtin_prefetch(xb + j0 + 128 + threadIdx.x, 0, 1);
        } else {
            const _Float16* xr = (const _Float16*)xv +
                ((size_t)b * NPTS + j0 + threadIdx.x) * (STRIDE ? STRIDE : 1);
#pragma unroll
            for (int c = 0; c < C; ++c)
                tile[threadIdx.x * CP + c] = (float)xr[c];
            if (j0 + 128 < NPTS)
                __builtin_prefetch(xr + 128 * (STRIDE ? STRIDE : 1), 0, 1);
        }
        __syncthreads();
        for (int jj = 0; jj < 128; ++jj) {
            float d = 0.0f;
#pragma unroll
            for (int c = 0; c < C; ++c) {
                float t = q[c] - tile[jj * CP + c];
                d -= t * t;
            }
            if (d > bd[KNN - 1]) {  // insertion into descending top-K
                float cd = d;
                int ci = j0 + jj;
#pragma unroll
                for (int k = 0; k < KNN; ++k) {
                    if (cd > bd[k]) {
                        float tf = bd[k]; bd[k] = cd; cd = tf;
                        int ti = bi[k];  bi[k] = ci; ci = ti;
                    }
                }
            }
        }
    }
    int* op = idx + ((size_t)b * NPTS + n) * KNN;
#pragma unroll
    for (int k = 0; k < KNN; ++k) op[k] = bi[k];
}

// ---------------------------------------------------------------------------
// Fused EdgeConv block: gather K neighbors for PT points, build edge features
// [diff | center] (2*CIN ch, padded to KPAD1), run two 1x1-conv layers as
// f16 WMMA GEMMs with folded-BN + LeakyReLU epilogues, then max over K via
// LDS u32-max with order-preserving float encoding. Output is f16 point-major
// [b][n][CPADOUT] so downstream stages can gather contiguously / async-copy.
//   A (weights):  [Mpad][Kpad] row-major f16; lane l loads row l%16,
//                 K (l/16)*16..+15  -> one contiguous 32B fragment.
//   B (acts):     LDS [edge][Kpad]; lane l is column l%16, same K split.
//   D:            lane l, vgpr r -> row r + 8*(l/16), col l%16.
// ---------------------------------------------------------------------------
template <int CIN, int INSTRIDE, int KPAD1, int M1T, int KPAD2, int M2,
          int M2T, int PT, int CPADOUT>
__global__ __launch_bounds__(256) void edge_mlp_kernel(
    const float* __restrict__ xf, const _Float16* __restrict__ xh,
    const int* __restrict__ idx, const _Float16* __restrict__ w1,
    const float* __restrict__ s1, const float* __restrict__ t1,
    const _Float16* __restrict__ w2, const float* __restrict__ s2,
    const float* __restrict__ t2, _Float16* __restrict__ xout) {
    constexpr int EDGES = PT * KNN;   // 80
    constexpr int NT = EDGES / 16;    // 5 column tiles
    constexpr int NW = 8;             // 8 wave32 per block
    struct __align__(32) Smem {
        _Float16 y1[EDGES * KPAD2];
        union {                       // feat is dead after GEMM1 -> reuse
            _Float16 feat[EDGES * KPAD1];
            unsigned omax[M2 * PT];
        } u;
    };
    __shared__ Smem sm;
    const int tid = threadIdx.x;
    const int lane = tid & 31;
    const int wave = tid >> 5;
    const int b = blockIdx.y;
    const int n0 = blockIdx.x * PT;

    // zero feature pad and y1 K-pad rows
    for (int i = tid; i < EDGES * KPAD1; i += 256) sm.u.feat[i] = (_Float16)0.0f;
    constexpr int YPAD = KPAD2 - M1T * 16;
    for (int i = tid; i < EDGES * YPAD; i += 256) {
        int e = i / YPAD;
        int c = M1T * 16 + (i - e * YPAD);
        sm.y1[e * KPAD2 + c] = (_Float16)0.0f;
    }
    __syncthreads();

    // gather + build edge features
    for (int e = tid; e < EDGES; e += 256) {
        int p = e / KNN, kk = e - p * KNN;
        int n = n0 + p;
        int nb = idx[((size_t)b * NPTS + n) * KNN + kk];
        if (INSTRIDE == 0) {
            const float* xb = xf + (size_t)b * CIN * NPTS;
#pragma unroll
            for (int c = 0; c < CIN; ++c) {
                float ctr = xb[c * NPTS + n];
                float nbr = xb[c * NPTS + nb];
                sm.u.feat[e * KPAD1 + c] = (_Float16)(nbr - ctr);
                sm.u.feat[e * KPAD1 + CIN + c] = (_Float16)ctr;
            }
        } else {
            constexpr int ST = INSTRIDE ? INSTRIDE : 1;
            const _Float16* xc = xh + ((size_t)b * NPTS + n) * ST;
            const _Float16* xn = xh + ((size_t)b * NPTS + nb) * ST;
#pragma unroll
            for (int c = 0; c < CIN; ++c) {
                float ctr = (float)xc[c];
                float nbr = (float)xn[c];
                sm.u.feat[e * KPAD1 + c] = (_Float16)(nbr - ctr);
                sm.u.feat[e * KPAD1 + CIN + c] = (_Float16)ctr;
            }
        }
    }
    __syncthreads();

    // ---- GEMM1: y1 = leaky(s1 .* (W1 @ feat) + t1), stored f16 ----
    for (int t = wave; t < M1T * NT; t += NW) {
        int mt = t / NT, nt = t - mt * NT;
        int mrow = mt * 16 + (lane & 15);
        int ecol = nt * 16 + (lane & 15);
        int koff = (lane >> 4) * 16;
        v8f acc = {0, 0, 0, 0, 0, 0, 0, 0};
#pragma unroll
        for (int ks = 0; ks < KPAD1 / 32; ++ks) {
            v16h a = *reinterpret_cast<const v16h*>(w1 + mrow * KPAD1 + ks * 32 + koff);
            v16h bm = *reinterpret_cast<const v16h*>(&sm.u.feat[ecol * KPAD1 + ks * 32 + koff]);
            acc = __builtin_amdgcn_wmma_f32_16x16x32_f16(false, a, false, bm,
                                                         (short)0, acc, false, false);
        }
#pragma unroll
        for (int r = 0; r < 8; ++r) {
            int m = mt * 16 + r + 8 * (lane >> 4);
            float y = s1[m] * acc[r] + t1[m];
            y = fmaxf(y, 0.2f * y);  // LeakyReLU(0.2)
            sm.y1[ecol * KPAD2 + m] = (_Float16)y;
        }
    }
    __syncthreads();

    // init max buffer (overlays feat); encoded 0 == -inf sentinel
    for (int i = tid; i < M2 * PT; i += 256) sm.u.omax[i] = 0u;
    __syncthreads();

    // ---- GEMM2 + max over K neighbors ----
    for (int t = wave; t < M2T * NT; t += NW) {
        int mt = t / NT, nt = t - mt * NT;
        int mrow = mt * 16 + (lane & 15);
        int ecol = nt * 16 + (lane & 15);
        int koff = (lane >> 4) * 16;
        v8f acc = {0, 0, 0, 0, 0, 0, 0, 0};
#pragma unroll
        for (int ks = 0; ks < KPAD2 / 32; ++ks) {
            v16h a = *reinterpret_cast<const v16h*>(w2 + mrow * KPAD2 + ks * 32 + koff);
            v16h bm = *reinterpret_cast<const v16h*>(&sm.y1[ecol * KPAD2 + ks * 32 + koff]);
            acc = __builtin_amdgcn_wmma_f32_16x16x32_f16(false, a, false, bm,
                                                         (short)0, acc, false, false);
        }
        int p = ecol / KNN;
#pragma unroll
        for (int r = 0; r < 8; ++r) {
            int m = mt * 16 + r + 8 * (lane >> 4);
            if (m < M2) {
                float y = s2[m] * acc[r] + t2[m];
                y = fmaxf(y, 0.2f * y);
                atomicMax(&sm.u.omax[m * PT + p], encf(y));  // ds_max_u32
            }
        }
    }
    __syncthreads();

    // write f16 point-major output (pad channels zeroed)
    for (int o = tid; o < CPADOUT * PT; o += 256) {
        int m = o / PT, p = o - m * PT;
        float v = (m < M2) ? decf(sm.u.omax[m * PT + p]) : 0.0f;
        xout[((size_t)b * NPTS + n0 + p) * CPADOUT + m] = (_Float16)v;
    }
}

// ---------------------------------------------------------------------------
// Head MLP: h = [x1 (48, padded) | x2 (512)] -> 531 -> 512 over 16-pt tiles.
// h staging is a pure f16 byte copy -> CDNA5 async global->LDS loads
// (GLOBAL_LOAD_ASYNC_TO_LDS_B128, ASYNCcnt) when the toolchain exposes them.
// ---------------------------------------------------------------------------
template <int KPAD1, int M1T, int KPAD2, int M2, int M2T, int X1P, int X2C>
__global__ __launch_bounds__(256) void mlp3_kernel(
    const _Float16* __restrict__ x1, const _Float16* __restrict__ x2,
    const _Float16* __restrict__ w1, const float* __restrict__ s1,
    const float* __restrict__ t1, const _Float16* __restrict__ w2,
    const float* __restrict__ s2, const float* __restrict__ t2,
    float* __restrict__ out) {
    constexpr int PT = 16;
    constexpr int NW = 8;
    struct __align__(32) Smem {
        _Float16 h[PT * KPAD1];
        _Float16 y1[PT * KPAD2];
    };
    __shared__ Smem sm;
    const int tid = threadIdx.x, lane = tid & 31, wave = tid >> 5;
    const int b = blockIdx.y;
    const int n0 = blockIdx.x * PT;

    // zero tail columns [X1P+X2C, KPAD1)
    constexpr int TAIL = KPAD1 - X1P - X2C;
    for (int i = tid; i < PT * TAIL; i += 256) {
        int p = i / TAIL, c = X1P + X2C + (i - p * TAIL);
        sm.h[p * KPAD1 + c] = (_Float16)0.0f;
    }

#if __has_builtin(__builtin_amdgcn_global_load_async_to_lds_b128)
    // async byte-copy: per point, 6 B128 from x1 row + 64 B128 from x2 row
    constexpr int J1 = (X1P * 2) / 16;        // 6
    constexpr int JT = J1 + (X2C * 2) / 16;   // 70
    for (int i = tid; i < PT * JT; i += 256) {
        int p = i / JT, j = i - p * JT;
        const _Float16* src =
            (j < J1) ? x1 + ((size_t)b * NPTS + n0 + p) * X1P + j * 8
                     : x2 + ((size_t)b * NPTS + n0 + p) * X2C + (j - J1) * 8;
        _Float16* dst = &sm.h[p * KPAD1 + j * 8];
        __builtin_amdgcn_global_load_async_to_lds_b128(
            (__attribute__((address_space(1))) i32x4*)src,
            (__attribute__((address_space(3))) i32x4*)dst, 0, 0);
    }
#if __has_builtin(__builtin_amdgcn_s_wait_asynccnt)
    __builtin_amdgcn_s_wait_asynccnt(0);
#else
    asm volatile("s_wait_asynccnt 0x0" ::: "memory");
#endif
#else
    for (int i = tid; i < PT * (X1P + X2C); i += 256) {
        int p = i / (X1P + X2C), c = i - p * (X1P + X2C);
        sm.h[p * KPAD1 + c] =
            (c < X1P) ? x1[((size_t)b * NPTS + n0 + p) * X1P + c]
                      : x2[((size_t)b * NPTS + n0 + p) * X2C + (c - X1P)];
    }
#endif
    __syncthreads();

    for (int t = wave; t < M1T; t += NW) {  // single 16-col tile
        int mrow = t * 16 + (lane & 15);
        int pcol = lane & 15;
        int koff = (lane >> 4) * 16;
        v8f acc = {0, 0, 0, 0, 0, 0, 0, 0};
#pragma unroll
        for (int ks = 0; ks < KPAD1 / 32; ++ks) {
            v16h a = *reinterpret_cast<const v16h*>(w1 + mrow * KPAD1 + ks * 32 + koff);
            v16h bm = *reinterpret_cast<const v16h*>(&sm.h[pcol * KPAD1 + ks * 32 + koff]);
            acc = __builtin_amdgcn_wmma_f32_16x16x32_f16(false, a, false, bm,
                                                         (short)0, acc, false, false);
        }
#pragma unroll
        for (int r = 0; r < 8; ++r) {
            int m = t * 16 + r + 8 * (lane >> 4);
            float y = s1[m] * acc[r] + t1[m];
            y = fmaxf(y, 0.2f * y);
            sm.y1[pcol * KPAD2 + m] = (_Float16)y;
        }
    }
    __syncthreads();

    for (int t = wave; t < M2T; t += NW) {
        int mrow = t * 16 + (lane & 15);
        int pcol = lane & 15;
        int koff = (lane >> 4) * 16;
        v8f acc = {0, 0, 0, 0, 0, 0, 0, 0};
#pragma unroll
        for (int ks = 0; ks < KPAD2 / 32; ++ks) {
            v16h a = *reinterpret_cast<const v16h*>(w2 + mrow * KPAD2 + ks * 32 + koff);
            v16h bm = *reinterpret_cast<const v16h*>(&sm.y1[pcol * KPAD2 + ks * 32 + koff]);
            acc = __builtin_amdgcn_wmma_f32_16x16x32_f16(false, a, false, bm,
                                                         (short)0, acc, false, false);
        }
#pragma unroll
        for (int r = 0; r < 8; ++r) {
            int m = t * 16 + r + 8 * (lane >> 4);
            if (m < M2) {
                float y = s2[m] * acc[r] + t2[m];
                y = fmaxf(y, 0.2f * y);
                out[((size_t)b * M2 + m) * NPTS + n0 + pcol] = y;
            }
        }
    }
}

// ---------------------------------------------------------------------------
// Host orchestration
// ---------------------------------------------------------------------------
extern "C" void kernel_launch(void* const* d_in, const int* in_sizes, int n_in,
                              void* d_out, int out_size, void* d_ws,
                              size_t ws_size, hipStream_t stream) {
    (void)out_size; (void)ws_size;
    const int N = NPTS, B = BATCH;

    // layer order: c1[0], c1[1], c2[0], c2[1], c3[0], c3[1]
    static const int LM[6]  = {15, 39, 199, 512, 531, 512};
    static const int LK[6]  = {6, 15, 78, 199, 551, 531};
    static const int LMP[6] = {16, 48, 208, 512, 544, 512};
    static const int LKP[6] = {32, 32, 96, 224, 576, 544};
    // c3[0] consumes the concat in [x1 padded to 48 | x2] layout
    static const int LSP[6] = {32, 32, 96, 224, 39, 544};  // ksplit
    static const int LSH[6] = {0, 0, 0, 0, 9, 0};          // kshift

    // Locate x and detect leaf ordering within each 6-entry layer group.
    int xIdx = 0, g0 = 1;
    if (n_in >= 37 && in_sizes[0] != B * 3 * N) { xIdx = n_in - 1; g0 = 0; }
    const float* x = (const float*)d_in[xIdx];

    const float *W[6], *Bb[6], *Ga[6], *Be[6], *Mu[6], *Va[6];
    for (int l = 0; l < 6; ++l) {
        int base = g0 + 6 * l;
        if (in_sizes[base] == LM[l] * LK[l]) {
            // insertion order: w, b, gamma, beta, mean, var
            W[l]  = (const float*)d_in[base + 0];
            Bb[l] = (const float*)d_in[base + 1];
            Ga[l] = (const float*)d_in[base + 2];
            Be[l] = (const float*)d_in[base + 3];
            Mu[l] = (const float*)d_in[base + 4];
            Va[l] = (const float*)d_in[base + 5];
        } else {
            // sorted leaves: b, beta, gamma, mean, var, w
            Bb[l] = (const float*)d_in[base + 0];
            Be[l] = (const float*)d_in[base + 1];
            Ga[l] = (const float*)d_in[base + 2];
            Mu[l] = (const float*)d_in[base + 3];
            Va[l] = (const float*)d_in[base + 4];
            W[l]  = (const float*)d_in[base + 5];
        }
    }

    // Workspace carve-out (256B aligned)
    char* ws = (char*)d_ws;
    size_t off = 0;
    auto take = [&](size_t bytes) -> char* {
        off = (off + 255) & ~(size_t)255;
        char* p = ws + off;
        off += bytes;
        return p;
    };
    int* idx1 = (int*)take((size_t)B * N * KNN * 4);
    int* idx2 = (int*)take((size_t)B * N * KNN * 4);
    _Float16* x1h = (_Float16*)take((size_t)B * N * 48 * 2);   // [b][n][48]
    _Float16* x2h = (_Float16*)take((size_t)B * N * 512 * 2);  // [b][n][512]
    _Float16* Wh[6];
    float *Sc[6], *Tc[6];
    for (int l = 0; l < 6; ++l) {
        Wh[l] = (_Float16*)take((size_t)LMP[l] * LKP[l] * 2);
        Sc[l] = (float*)take((size_t)LMP[l] * 4);
        Tc[l] = (float*)take((size_t)LMP[l] * 4);
    }

    // 1. BN folding + f16 weight conversion (with c3 concat column remap)
    for (int l = 0; l < 6; ++l)
        prep_layer_kernel<<<dim3(32), dim3(256), 0, stream>>>(
            W[l], Bb[l], Ga[l], Be[l], Mu[l], Va[l], LM[l], LK[l], LMP[l],
            LKP[l], LSP[l], LSH[l], Wh[l], Sc[l], Tc[l]);

    // 2. KNN on x (C=3, f32 channel-major) -> edge MLP 6->15->39 -> x1 (f16)
    knn_kernel<3, 0><<<dim3(N / 128, B), dim3(128), 0, stream>>>(x, idx1);
    edge_mlp_kernel<3, 0, 32, 1, 32, 39, 3, 8, 48>
        <<<dim3(N / 8, B), dim3(256), 0, stream>>>(x, nullptr, idx1, Wh[0],
                                                   Sc[0], Tc[0], Wh[1], Sc[1],
                                                   Tc[1], x1h);

    // 3. KNN on x1 (C=39, f16 point-major) -> edge MLP 78->199->512 -> x2
    knn_kernel<39, 48><<<dim3(N / 128, B), dim3(128), 0, stream>>>(x1h, idx2);
    edge_mlp_kernel<39, 48, 96, 13, 224, 512, 32, 8, 512>
        <<<dim3(N / 8, B), dim3(256), 0, stream>>>(nullptr, x1h, idx2, Wh[2],
                                                   Sc[2], Tc[2], Wh[3], Sc[3],
                                                   Tc[3], x2h);

    // 4. concat(x1,x2) -> 551->531->512 -> d_out (f32, channel-major)
    mlp3_kernel<576, 34, 544, 512, 32, 48, 512>
        <<<dim3(N / 16, B), dim3(256), 0, stream>>>(x1h, x2h, Wh[4], Sc[4],
                                                    Tc[4], Wh[5], Sc[5], Tc[5],
                                                    (float*)d_out);
}
